// GruGatNets_64424509440156
// MI455X (gfx1250) — matrix operand order repaired and checked by reading, not compiled, not run on hardware
//
#include <hip/hip_runtime.h>

typedef __bf16 bf16;
typedef __attribute__((ext_vector_type(4)))  bf16  bf16x4;
typedef __attribute__((ext_vector_type(8)))  bf16  bf16x8;
typedef __attribute__((ext_vector_type(16))) bf16  v16bf;
typedef __attribute__((ext_vector_type(8)))  float v8f;

#define H 128

// ---------------- WMMA helpers (gfx1250 layouts) ----------------

__device__ __forceinline__ v8f wmma_bf16(v16bf a, v16bf b, v8f c) {
    return __builtin_amdgcn_wmma_f32_16x16x32_bf16(
        false, a, false, b, (short)0, c, false, false);
}

// A-matrix 16x32 bf16 fragment: lane holds row (lane&15); K pairs at
// 8*g..8*g+7 and 16+8*g..16+8*g+7 (g = lane>>4), contiguous 16B runs.
__device__ __forceinline__ v16bf frag_A_bf16(const bf16* __restrict__ base,
                                             int k0, int lane) {
    int r = lane & 15, g = lane >> 4;
    const bf16* p = base + (size_t)r * H + k0 + 8 * g;
    bf16x8 lo = *(const bf16x8*)p;
    bf16x8 hi = *(const bf16x8*)(p + 16);
    v16bf f;
#pragma unroll
    for (int i = 0; i < 8; i++) { f[i] = lo[i]; f[i + 8] = hi[i]; }
    return f;
}

// Same A layout but source is f32 (convert to bf16 on load).
__device__ __forceinline__ v16bf frag_A_f32(const float* __restrict__ base,
                                            int k0, int lane) {
    int r = lane & 15, g = lane >> 4;
    const float* p = base + (size_t)r * H + k0 + 8 * g;
    v16bf f;
#pragma unroll
    for (int i = 0; i < 8; i++) { f[i] = (bf16)p[i]; f[i + 8] = (bf16)p[i + 16]; }
    return f;
}

// B-matrix 32x16 bf16 fragment. Our GEMM is h @ W.T, so B[k][n] = W[n][k]:
// lane holds W row n=(lane&15); K at k0+8*g..+7 and k0+16+8*g..+7 (same as A).
__device__ __forceinline__ v16bf frag_B_bf16(const bf16* __restrict__ Wrow0,
                                             int k0, int lane) {
    int n = lane & 15, g = lane >> 4;
    const bf16* p = Wrow0 + (size_t)n * H + k0 + 8 * g;
    bf16x8 lo = *(const bf16x8*)p;
    bf16x8 hi = *(const bf16x8*)(p + 16);
    v16bf f;
#pragma unroll
    for (int i = 0; i < 8; i++) { f[i] = lo[i]; f[i + 8] = hi[i]; }
    return f;
}

// ---------------- generic small kernels ----------------

__global__ void k_zero32(unsigned* __restrict__ p, long long n) {
    long long i = (long long)blockIdx.x * blockDim.x + threadIdx.x;
    if (i < n) p[i] = 0u;
}

__global__ void k_cvt_bf16(const float* __restrict__ in, bf16* __restrict__ out, int n) {
    int i = blockIdx.x * blockDim.x + threadIdx.x;
    if (i < n) out[i] = (bf16)in[i];
}

// gather rows of res into h (f32) + hbf (bf16); zero padding rows
__global__ void k_gather(const float* __restrict__ res, const int* __restrict__ nid,
                         float* __restrict__ h, bf16* __restrict__ hbf,
                         int Nt, long long total) {
    long long idx = (long long)blockIdx.x * blockDim.x + threadIdx.x;
    if (idx >= total) return;
    int i = (int)(idx >> 7), f = (int)(idx & 127);
    float v = (i < Nt) ? res[(size_t)nid[i] * H + f] : 0.0f;
    h[idx]  = v;
    hbf[idx] = (bf16)v;
}

__global__ void k_scatter(float* __restrict__ res, const int* __restrict__ nid,
                          const float* __restrict__ h, long long total) {
    long long idx = (long long)blockIdx.x * blockDim.x + threadIdx.x;
    if (idx >= total) return;
    int i = (int)(idx >> 7), f = (int)(idx & 127);
    res[(size_t)nid[i] * H + f] = h[idx];
}

// ---------------- GEMM: out[m][n] = A[m][:] . W[n][:] (+bias) ----------------
// grid = (padNt/16, 8), block = 32 (one wave). A: [padNt,128] bf16, W: [128,128] bf16.
// MODE 0: bf16 output with bias; MODE 1: f32 output, no bias.
template <int MODE>
__global__ void __launch_bounds__(32)
k_gemm128(const bf16* __restrict__ A, const bf16* __restrict__ W,
          const float* __restrict__ bias, bf16* __restrict__ outB,
          float* __restrict__ outF) {
    int lane = threadIdx.x & 31;
    int rt = blockIdx.x, ct = blockIdx.y;
    const bf16* Arow = A + (size_t)rt * 16 * H;
    const bf16* Wrow = W + (size_t)ct * 16 * H;
    v8f acc = {};
#pragma unroll
    for (int kc = 0; kc < 4; kc++) {
        v16bf af  = frag_A_bf16(Arow, kc * 32, lane);
        v16bf bf_ = frag_B_bf16(Wrow, kc * 32, lane);
        acc = wmma_bf16(af, bf_, acc);
    }
    int col = ct * 16 + (lane & 15);
    int row0 = rt * 16 + 8 * (lane >> 4);
    if (MODE == 0) {
        float bv = bias[col];
#pragma unroll
        for (int j = 0; j < 8; j++)
            outB[(size_t)(row0 + j) * H + col] = (bf16)(acc[j] + bv);
    } else {
#pragma unroll
        for (int j = 0; j < 8; j++)
            outF[(size_t)(row0 + j) * H + col] = acc[j];
    }
}

// ---------------- edge scatter: a[dst] += m[src], 4 feats/thread ----------------
__global__ void k_edge_scatter(const bf16* __restrict__ m, const int* __restrict__ src,
                               const int* __restrict__ dst, float* __restrict__ a,
                               long long total) {
    long long idx = (long long)blockIdx.x * blockDim.x + threadIdx.x;
    if (idx >= total) return;
    int e = (int)(idx >> 5), f4 = (int)(idx & 31) * 4;
    bf16x4 mv = *(const bf16x4*)(m + (size_t)src[e] * H + f4);
    float* ap = a + (size_t)dst[e] * H + f4;
#pragma unroll
    for (int k = 0; k < 4; k++) atomicAdd(ap + k, (float)mv[k]);
}

// ---------------- fused GRU update ----------------
// grid = padNt/16 blocks, 128 threads (4 waves). Wave w owns cols [32w,32w+32).
__global__ void __launch_bounds__(128)
k_gru(const float* __restrict__ a, float* __restrict__ h, bf16* __restrict__ hbf,
      const bf16* __restrict__ Wih, const bf16* __restrict__ Whh,
      const float* __restrict__ bih, const float* __restrict__ bhh, int Nt) {
    int lane = threadIdx.x & 31;
    int w = threadIdx.x >> 5;
    int rt = blockIdx.x;
    int b = w * 32;
    const float* Arow = a   + (size_t)rt * 16 * H;
    const bf16*  Hrow = hbf + (size_t)rt * 16 * H;

    v16bf afrag[4], hfrag[4];
#pragma unroll
    for (int kc = 0; kc < 4; kc++) {
        afrag[kc] = frag_A_f32(Arow, kc * 32, lane);
        hfrag[kc] = frag_A_bf16(Hrow, kc * 32, lane);
    }
    __syncthreads();   // all waves finished reading h/hbf before any write

    v8f ir[2], hr[2], iz[2], hz[2];
#pragma unroll
    for (int t = 0; t < 2; t++)
#pragma unroll
        for (int i = 0; i < 8; i++) { ir[t][i] = 0.f; hr[t][i] = 0.f; iz[t][i] = 0.f; hz[t][i] = 0.f; }

#pragma unroll
    for (int kc = 0; kc < 4; kc++) {
#pragma unroll
        for (int t = 0; t < 2; t++) {
            int c0 = b + 16 * t;
            ir[t] = wmma_bf16(afrag[kc], frag_B_bf16(Wih + (size_t)c0 * H,         kc * 32, lane), ir[t]);
            hr[t] = wmma_bf16(hfrag[kc], frag_B_bf16(Whh + (size_t)c0 * H,         kc * 32, lane), hr[t]);
            iz[t] = wmma_bf16(afrag[kc], frag_B_bf16(Wih + (size_t)(128 + c0) * H, kc * 32, lane), iz[t]);
            hz[t] = wmma_bf16(hfrag[kc], frag_B_bf16(Whh + (size_t)(128 + c0) * H, kc * 32, lane), hz[t]);
        }
    }

    float rr[2][8], zz[2][8];
#pragma unroll
    for (int t = 0; t < 2; t++) {
        int col = b + 16 * t + (lane & 15);
#pragma unroll
        for (int j = 0; j < 8; j++) {
            float xr = ir[t][j] + bih[col] + hr[t][j] + bhh[col];
            float xz = iz[t][j] + bih[128 + col] + hz[t][j] + bhh[128 + col];
            rr[t][j] = 1.0f / (1.0f + expf(-xr));
            zz[t][j] = 1.0f / (1.0f + expf(-xz));
        }
    }

    v8f inn[2], hn[2];
#pragma unroll
    for (int t = 0; t < 2; t++)
#pragma unroll
        for (int i = 0; i < 8; i++) { inn[t][i] = 0.f; hn[t][i] = 0.f; }

#pragma unroll
    for (int kc = 0; kc < 4; kc++) {
#pragma unroll
        for (int t = 0; t < 2; t++) {
            int c0 = 256 + b + 16 * t;
            inn[t] = wmma_bf16(afrag[kc], frag_B_bf16(Wih + (size_t)c0 * H, kc * 32, lane), inn[t]);
            hn[t]  = wmma_bf16(hfrag[kc], frag_B_bf16(Whh + (size_t)c0 * H, kc * 32, lane), hn[t]);
        }
    }

#pragma unroll
    for (int t = 0; t < 2; t++) {
        int col = b + 16 * t + (lane & 15);
#pragma unroll
        for (int j = 0; j < 8; j++) {
            int row = rt * 16 + j + 8 * (lane >> 4);
            float ng = tanhf(inn[t][j] + bih[256 + col] + rr[t][j] * (hn[t][j] + bhh[256 + col]));
            float hold = h[(size_t)row * H + col];
            float hv = (1.0f - zz[t][j]) * ng + zz[t][j] * hold;
            if (row < Nt) {
                h[(size_t)row * H + col] = hv;
                hbf[(size_t)row * H + col] = (bf16)hv;
            }
        }
    }
}

// ---------------- GAT ----------------

__device__ __forceinline__ unsigned f32_ord(float x) {
    unsigned u = __float_as_uint(x);
    return (u & 0x80000000u) ? ~u : (u | 0x80000000u);
}
__device__ __forceinline__ float f32_unord(unsigned u) {
    return __uint_as_float((u & 0x80000000u) ? (u ^ 0x80000000u) : ~u);
}

// one wave per node: el = z.attn_l, er = z.attn_r
__global__ void k_attn(const float* __restrict__ z, const float* __restrict__ al,
                       const float* __restrict__ ar, float* __restrict__ el,
                       float* __restrict__ er, int Nd) {
    int wid = (int)(((long long)blockIdx.x * blockDim.x + threadIdx.x) >> 5);
    int lane = threadIdx.x & 31;
    if (wid >= Nd) return;
    const float* zr = z + (size_t)wid * H;
    float sl = 0.f, sr = 0.f;
#pragma unroll
    for (int k = 0; k < 4; k++) {
        float zv = zr[lane * 4 + k];
        sl += zv * al[lane * 4 + k];
        sr += zv * ar[lane * 4 + k];
    }
#pragma unroll
    for (int o = 16; o > 0; o >>= 1) { sl += __shfl_down(sl, o); sr += __shfl_down(sr, o); }
    if (lane == 0) { el[wid] = sl; er[wid] = sr; }
}

__global__ void k_edge_emax(const float* __restrict__ el, const float* __restrict__ er,
                            const int* __restrict__ src, const int* __restrict__ dst,
                            float* __restrict__ ebuf, unsigned* __restrict__ emaxU, int Ed) {
    int e = blockIdx.x * blockDim.x + threadIdx.x;
    if (e >= Ed) return;
    float v = el[src[e]] + er[dst[e]];
    v = (v > 0.0f) ? v : 0.2f * v;        // leaky_relu, slope 0.2
    ebuf[e] = v;
    atomicMax(&emaxU[dst[e]], f32_ord(v));
}

__global__ void k_edge_ee(float* __restrict__ ebuf, const int* __restrict__ dst,
                          const unsigned* __restrict__ emaxU, float* __restrict__ denom, int Ed) {
    int e = blockIdx.x * blockDim.x + threadIdx.x;
    if (e >= Ed) return;
    float ee = expf(ebuf[e] - f32_unord(emaxU[dst[e]]));
    ebuf[e] = ee;
    atomicAdd(&denom[dst[e]], ee);
}

// 4 feats/thread; thread group of 32 covers one edge's 128 feats
__global__ void k_edge_alpha(const float* __restrict__ ebuf, const float* __restrict__ denom,
                             const int* __restrict__ src, const int* __restrict__ dst,
                             const float* __restrict__ z, float* __restrict__ rst,
                             float* __restrict__ alpha_out, long long total) {
    long long idx = (long long)blockIdx.x * blockDim.x + threadIdx.x;
    if (idx >= total) return;
    int e = (int)(idx >> 5), f4 = (int)(idx & 31) * 4;
    float al = ebuf[e] / denom[dst[e]];
    if (f4 == 0) alpha_out[e] = al;
    const float4 zv = *(const float4*)(z + (size_t)src[e] * H + f4);
    float* rp = rst + (size_t)dst[e] * H + f4;
    atomicAdd(rp + 0, al * zv.x);
    atomicAdd(rp + 1, al * zv.y);
    atomicAdd(rp + 2, al * zv.z);
    atomicAdd(rp + 3, al * zv.w);
}

__global__ void k_gat_fin(const float* __restrict__ rst, const float* __restrict__ gbias,
                          const int* __restrict__ nid, float* __restrict__ res, long long total) {
    long long idx = (long long)blockIdx.x * blockDim.x + threadIdx.x;
    if (idx >= total) return;
    int i = (int)(idx >> 7), f = (int)(idx & 127);
    float v = rst[idx] + gbias[f];
    v = (v > 0.0f) ? v : expm1f(v);       // ELU(alpha=1)
    res[(size_t)nid[i] * H + f] = v;
}

// ---------------- pooling ----------------

__global__ void k_gate(const float* __restrict__ res, const float* __restrict__ Wg,
                       const float* __restrict__ bg, float* __restrict__ gate,
                       unsigned* __restrict__ gmaxU, int N) {
    int wid = (int)(((long long)blockIdx.x * blockDim.x + threadIdx.x) >> 5);
    int lane = threadIdx.x & 31;
    if (wid >= N) return;
    const float* rr = res + (size_t)wid * H;
    float s = 0.f;
#pragma unroll
    for (int k = 0; k < 4; k++) s += rr[lane * 4 + k] * Wg[lane * 4 + k];
#pragma unroll
    for (int o = 16; o > 0; o >>= 1) s += __shfl_down(s, o);
    if (lane == 0) {
        s += bg[0];
        gate[wid] = s;
        atomicMax(gmaxU, f32_ord(s));
    }
}

__global__ void k_gsum(const float* __restrict__ gate, const unsigned* __restrict__ gmaxU,
                       float* __restrict__ gsum, int N) {
    __shared__ float sh[256];
    float gm = f32_unord(*gmaxU);
    float s = 0.f;
    for (int i = blockIdx.x * blockDim.x + threadIdx.x; i < N; i += gridDim.x * blockDim.x)
        s += expf(gate[i] - gm);
    sh[threadIdx.x] = s;
    __syncthreads();
    for (int o = 128; o > 0; o >>= 1) {
        if ((int)threadIdx.x < o) sh[threadIdx.x] += sh[threadIdx.x + o];
        __syncthreads();
    }
    if (threadIdx.x == 0) atomicAdd(gsum, sh[0]);
}

__global__ void k_gweights(float* __restrict__ gate, const unsigned* __restrict__ gmaxU,
                           const float* __restrict__ gsum, int N) {
    int i = blockIdx.x * blockDim.x + threadIdx.x;
    if (i < N) gate[i] = expf(gate[i] - f32_unord(*gmaxU)) / (*gsum);
}

__global__ void __launch_bounds__(128)
k_readout(const float* __restrict__ res, const float* __restrict__ gate,
          float* __restrict__ readout, int N) {
    int f = threadIdx.x;
    int i0 = blockIdx.x * 512;
    int i1 = (i0 + 512 < N) ? (i0 + 512) : N;
    float acc = 0.f;
    for (int i = i0; i < i1; i++) acc += gate[i] * res[(size_t)i * H + f];
    atomicAdd(&readout[f], acc);
}

__global__ void __launch_bounds__(128)
k_pred(const float* __restrict__ readout, const float* __restrict__ Wcls,
       const float* __restrict__ bcls, float* __restrict__ pred) {
    __shared__ float sh[128];
    int f = threadIdx.x;
    for (int c = 0; c < 2; c++) {
        sh[f] = readout[f] * Wcls[c * H + f];
        __syncthreads();
        for (int o = 64; o > 0; o >>= 1) {
            if (f < o) sh[f] += sh[f + o];
            __syncthreads();
        }
        if (f == 0) pred[c] = sh[0] + bcls[c];
        __syncthreads();
    }
}

// ---------------- host launch ----------------

extern "C" void kernel_launch(void* const* d_in, const int* in_sizes, int n_in,
                              void* d_out, int out_size, void* d_ws, size_t ws_size,
                              hipStream_t stream) {
    (void)n_in; (void)out_size; (void)ws_size;
    const int N = in_sizes[12] / H;
    const int PAD = (N + 15) & ~15;

    float* pred      = (float*)d_out;
    float* res       = pred + 2;
    float* readout   = res + (size_t)N * H;
    float* alpha_out = readout + H;

    const int Nd = in_sizes[9];
    const int Ed = in_sizes[10];

    // --- workspace carve ---
    char* ws = (char*)d_ws;
    size_t off = 0;
    auto carve = [&](size_t bytes) -> void* {
        void* p = ws + off;
        off += (bytes + 255) & ~(size_t)255;
        return p;
    };
    float*    hbuf  = (float*)carve((size_t)PAD * H * 4);
    bf16*     hbf   = (bf16*) carve((size_t)PAD * H * 2);
    bf16*     mbf   = (bf16*) carve((size_t)PAD * H * 2);
    float*    abuf  = (float*)carve((size_t)PAD * H * 4);  // GRU 'a', then GAT 'z'
    float*    rst   = (float*)carve((size_t)PAD * H * 4);
    float*    el    = (float*)carve((size_t)PAD * 4);
    float*    er    = (float*)carve((size_t)PAD * 4);
    float*    ebuf  = (float*)carve((size_t)Ed * 4);
    unsigned* emaxU = (unsigned*)carve((size_t)PAD * 4);
    float*    denom = (float*)carve((size_t)PAD * 4);
    float*    gate  = (float*)carve((size_t)PAD * 4);
    unsigned* gmaxU = (unsigned*)carve(4);
    float*    gsum  = (float*)carve(4);
    bf16* Wmb[3], *Wihb[3], *Whhb[3];
    for (int t = 0; t < 3; t++) {
        Wmb[t]  = (bf16*)carve((size_t)H * H * 2);
        Wihb[t] = (bf16*)carve((size_t)3 * H * H * 2);
        Whhb[t] = (bf16*)carve((size_t)3 * H * H * 2);
    }
    bf16* Wfcb = (bf16*)carve((size_t)H * H * 2);

    // --- weight conversion to bf16 ---
    for (int t = 0; t < 3; t++) {
        int base = 13 + 6 * t;
        k_cvt_bf16<<<(H * H + 255) / 256, 256, 0, stream>>>((const float*)d_in[base], Wmb[t], H * H);
        k_cvt_bf16<<<(3 * H * H + 255) / 256, 256, 0, stream>>>((const float*)d_in[base + 2], Wihb[t], 3 * H * H);
        k_cvt_bf16<<<(3 * H * H + 255) / 256, 256, 0, stream>>>((const float*)d_in[base + 4], Whhb[t], 3 * H * H);
    }
    k_cvt_bf16<<<(H * H + 255) / 256, 256, 0, stream>>>((const float*)d_in[31], Wfcb, H * H);

    // res = feats
    hipMemcpyAsync(res, d_in[12], (size_t)N * H * sizeof(float),
                   hipMemcpyDeviceToDevice, stream);
    k_zero32<<<1, 128, 0, stream>>>((unsigned*)readout, H);

    // --- three GRU blocks ---
    for (int t = 0; t < 3; t++) {
        const int* nid = (const int*)d_in[3 * t];
        const int* src = (const int*)d_in[3 * t + 1];
        const int* dst = (const int*)d_in[3 * t + 2];
        const int Nt = in_sizes[3 * t];
        const int Et = in_sizes[3 * t + 1];
        const int padNt = (Nt + 15) & ~15;
        const int base = 13 + 6 * t;
        const float* bm  = (const float*)d_in[base + 1];
        const float* bih = (const float*)d_in[base + 3];
        const float* bhh = (const float*)d_in[base + 5];

        long long rows = (long long)padNt * H;
        k_gather<<<(unsigned)((rows + 255) / 256), 256, 0, stream>>>(res, nid, hbuf, hbf, Nt, rows);

        for (int step = 0; step < 2; step++) {
            k_gemm128<0><<<dim3(padNt / 16, 8), 32, 0, stream>>>(hbf, Wmb[t], bm, mbf, nullptr);
            k_zero32<<<(unsigned)((rows + 255) / 256), 256, 0, stream>>>((unsigned*)abuf, rows);
            long long etot = (long long)Et * 32;
            k_edge_scatter<<<(unsigned)((etot + 255) / 256), 256, 0, stream>>>(mbf, src, dst, abuf, etot);
            k_gru<<<padNt / 16, 128, 0, stream>>>(abuf, hbuf, hbf, Wihb[t], Whhb[t], bih, bhh, Nt);
        }
        long long vrows = (long long)Nt * H;
        k_scatter<<<(unsigned)((vrows + 255) / 256), 256, 0, stream>>>(res, nid, hbuf, vrows);
    }

    // --- GATConv on remaining edge type ---
    {
        const int* nid = (const int*)d_in[9];
        const int* src = (const int*)d_in[10];
        const int* dst = (const int*)d_in[11];
        const int padNd = (Nd + 15) & ~15;
        long long rows = (long long)padNd * H;

        k_gather<<<(unsigned)((rows + 255) / 256), 256, 0, stream>>>(res, nid, hbuf, hbf, Nd, rows);
        k_gemm128<1><<<dim3(padNd / 16, 8), 32, 0, stream>>>(hbf, Wfcb, nullptr, nullptr, abuf); // z
        k_attn<<<(Nd + 3) / 4, 128, 0, stream>>>(abuf, (const float*)d_in[32], (const float*)d_in[33], el, er, Nd);
        k_zero32<<<(padNd + 255) / 256, 256, 0, stream>>>(emaxU, padNd);
        k_zero32<<<(padNd + 255) / 256, 256, 0, stream>>>((unsigned*)denom, padNd);
        k_edge_emax<<<(Ed + 255) / 256, 256, 0, stream>>>(el, er, src, dst, ebuf, emaxU, Ed);
        k_edge_ee<<<(Ed + 255) / 256, 256, 0, stream>>>(ebuf, dst, emaxU, denom, Ed);
        k_zero32<<<(unsigned)((rows + 255) / 256), 256, 0, stream>>>((unsigned*)rst, rows);
        long long etot = (long long)Ed * 32;
        k_edge_alpha<<<(unsigned)((etot + 255) / 256), 256, 0, stream>>>(ebuf, denom, src, dst, abuf, rst, alpha_out, etot);
        long long vrows = (long long)Nd * H;
        k_gat_fin<<<(unsigned)((vrows + 255) / 256), 256, 0, stream>>>(rst, (const float*)d_in[34], nid, res, vrows);
    }

    // --- global attention pooling + classifier ---
    k_zero32<<<1, 32, 0, stream>>>(gmaxU, 1);
    k_zero32<<<1, 32, 0, stream>>>((unsigned*)gsum, 1);
    k_gate<<<(N + 3) / 4, 128, 0, stream>>>(res, (const float*)d_in[35], (const float*)d_in[36], gate, gmaxU, N);
    k_gsum<<<128, 256, 0, stream>>>(gate, gmaxU, gsum, N);
    k_gweights<<<(N + 255) / 256, 256, 0, stream>>>(gate, gmaxU, gsum, N);
    k_readout<<<(N + 511) / 512, 128, 0, stream>>>(res, gate, readout, N);
    k_pred<<<1, 128, 0, stream>>>(readout, (const float*)d_in[37], (const float*)d_in[38], pred);
}
